// GNN_ActorCritic_63840393888429
// MI455X (gfx1250) — matrix operand order repaired and checked by reading, not compiled, not run on hardware
//
#include <hip/hip_runtime.h>

// ---------------- problem constants (from reference setup_inputs) ----------
#define V_N   1024
#define C_N   4096
#define A_N   4
#define M_N   256
#define FV    8
#define H     128
#define E_N   16
#define STEPS 8
#define XIN   272      // 2H + E  (actor input)
#define VCK   137      // H + 1 + FV (gru_v input K)

// ---------------- WMMA types ----------------------------------------------
typedef __bf16 bf16x16 __attribute__((ext_vector_type(16)));
typedef float  v8f     __attribute__((ext_vector_type(8)));

union Frag { bf16x16 v; uint4 u[2]; };

// pack two f32 -> packed bf16 dword (round-to-nearest-even)
#if defined(__has_builtin) && __has_builtin(__builtin_amdgcn_cvt_pk_bf16_f32)
typedef __bf16 bf16x2 __attribute__((ext_vector_type(2)));
__device__ __forceinline__ unsigned pack2bf(float a, float b) {
  bf16x2 p = __builtin_amdgcn_cvt_pk_bf16_f32(a, b);   // v_cvt_pk_bf16_f32
  return __builtin_bit_cast(unsigned, p);
}
#else
__device__ __forceinline__ unsigned pack2bf(float a, float b) {
  unsigned ua = __float_as_uint(a), ub = __float_as_uint(b);
  ua += 0x7FFFu + ((ua >> 16) & 1u);
  ub += 0x7FFFu + ((ub >> 16) & 1u);
  return (ua >> 16) | (ub & 0xFFFF0000u);
}
#endif

// ---------------- tiled WMMA GEMM ------------------------------------------
// C[M,N] = act( op(A) @ B + bias ), op(A)=A (TA=0, A is [M,K] lda) or
// op(A)=A^T (TA=1, A stored [K,M] lda). MASKED=1 applies the rank-1 edge mask
// mmask[M] (rows of op(A)) x kmask[K] multiplicatively to A while staging.
// Staging: each thread owns 16 consecutive K elements of one row/column ->
// coalesced global reads, packed bf16 conversion, two ds_store_b128 per
// matrix per thread (transpose absorbed by the thread mapping, no scatter).
// Double-buffered LDS with split load/store software pipeline.
#define BM 128
#define BN 128
#define BK 32
#define ASTR 40   // ushort row stride (80B: keeps 16B alignment, dodges banks)

struct StageRegs { float a[16]; float b[16]; };

template<int TA, int MASKED>
__device__ __forceinline__ void stage_load(StageRegs& rg,
    const float* __restrict__ A, const float* __restrict__ B,
    int M, int N, int K, int lda, int ldb,
    const float* __restrict__ mmask, const float* __restrict__ kmask,
    int bm0, int bn0, int kt, int tid)
{
  if (TA == 0) {           // A[M,K]: thread = one row r, 16 consecutive k
    int r = tid >> 1, kh = (tid & 1) * 16, gm = bm0 + r;
    int gmc = gm < M ? gm : M - 1;
    float rm = MASKED ? mmask[gmc] : 1.f;
    if (gm >= M) rm = 0.f;
    const float* Arow = A + (long)gmc * lda;
    for (int j = 0; j < 16; ++j) {
      int gk = kt + kh + j, gkc = gk < K ? gk : K - 1;
      float v = Arow[gkc];
      float km = MASKED ? kmask[gkc] : 1.f;
      rg.a[j] = (gk < K) ? v * km * rm : 0.f;
    }
  } else {                 // A[K,M]: thread = one column m, 16 consecutive k
    int m = tid & 127, kh = (tid >> 7) * 16, gm = bm0 + m;
    int gmc = gm < M ? gm : M - 1;
    float rm = MASKED ? mmask[gmc] : 1.f;
    if (gm >= M) rm = 0.f;
    for (int j = 0; j < 16; ++j) {
      int gk = kt + kh + j, gkc = gk < K ? gk : K - 1;
      float v = A[(long)gkc * lda + gmc];
      float km = MASKED ? kmask[gkc] : 1.f;
      rg.a[j] = (gk < K) ? v * km * rm : 0.f;
    }
  }
  {                        // B[K,N]: thread = one column n, 16 consecutive k
    int n = tid & 127, kh = (tid >> 7) * 16, gn = bn0 + n;
    int gnc = gn < N ? gn : N - 1;
    float nm = (gn < N) ? 1.f : 0.f;
    for (int j = 0; j < 16; ++j) {
      int gk = kt + kh + j, gkc = gk < K ? gk : K - 1;
      float v = B[(long)gkc * ldb + gnc];
      rg.b[j] = (gk < K) ? v * nm : 0.f;
    }
  }
}

__device__ __forceinline__ void store16(const float* s, unsigned short* dst) {
  uint4 w0 = { pack2bf(s[0], s[1]),  pack2bf(s[2], s[3]),
               pack2bf(s[4], s[5]),  pack2bf(s[6], s[7]) };
  uint4 w1 = { pack2bf(s[8], s[9]),  pack2bf(s[10], s[11]),
               pack2bf(s[12], s[13]), pack2bf(s[14], s[15]) };
  ((uint4*)dst)[0] = w0;   // ds_store_b128
  ((uint4*)dst)[1] = w1;   // ds_store_b128
}

template<int TA>
__device__ __forceinline__ void stage_store(const StageRegs& rg,
    unsigned short* __restrict__ As, unsigned short* __restrict__ Bs, int tid)
{
  if (TA == 0) { int r = tid >> 1,  kh = (tid & 1) * 16;  store16(rg.a, &As[r * ASTR + kh]); }
  else         { int m = tid & 127, kh = (tid >> 7) * 16; store16(rg.a, &As[m * ASTR + kh]); }
  int n = tid & 127, kh = (tid >> 7) * 16;
  store16(rg.b, &Bs[n * ASTR + kh]);
}

template<int TA, int MASKED>
__global__ __launch_bounds__(256)
void gemm_k(const float* __restrict__ A, const float* __restrict__ B,
            const float* __restrict__ bias, float* __restrict__ C,
            int M, int N, int K, int lda, int ldb, int ldc,
            const float* __restrict__ mmask, const float* __restrict__ kmask,
            int relu)
{
  __shared__ unsigned short As[2][BM * ASTR];   // As[buf][row][k] (bf16 bits)
  __shared__ unsigned short Bs[2][BN * ASTR];   // Bs[buf][n][k]  (K-transposed)
  const int tid = threadIdx.x;
  const int bm0 = blockIdx.x * BM, bn0 = blockIdx.y * BN;
  const int wid = tid >> 5, lane = tid & 31, lh = lane >> 4, l2 = lane & 15;
  const int wm = (wid & 3) * 32, wn = (wid >> 2) * 64;   // 8 waves: 4x2 grid

  v8f acc[2][4] = {};
  const int nk = (K + BK - 1) / BK;

  StageRegs rg;
  stage_load<TA, MASKED>(rg, A, B, M, N, K, lda, ldb, mmask, kmask, bm0, bn0, 0, tid);
  stage_store<TA>(rg, As[0], Bs[0], tid);
  int cur = 0;

  for (int it = 0; it < nk; ++it) {
    __syncthreads();   // buffer 'cur' staged & prior compute on cur^1 done
    // prefetch next tile into registers (vmem issued before this tile's math)
    if (it + 1 < nk)
      stage_load<TA, MASKED>(rg, A, B, M, N, K, lda, ldb, mmask, kmask,
                             bm0, bn0, (it + 1) * BK, tid);

    // ---- fragments (contiguous k = lh*16 + s in A and B), then 8 WMMAs ----
    Frag a[2], b[4];
    const unsigned short* Ab = As[cur];
    const unsigned short* Bb = Bs[cur];
    for (int fm = 0; fm < 2; ++fm) {
      int row = wm + fm * 16 + l2;
      a[fm].u[0] = *(const uint4*)&Ab[row * ASTR + lh * 16];       // K=lh*16..+7
      a[fm].u[1] = *(const uint4*)&Ab[row * ASTR + lh * 16 + 8];   // K=lh*16+8..+15
    }
    for (int fn = 0; fn < 4; ++fn) {
      int n = wn + fn * 16 + l2;
      const uint4* p0 = (const uint4*)&Bb[n * ASTR + lh * 16];     // K=lh*16..+15
      b[fn].u[0] = p0[0]; b[fn].u[1] = p0[1];
    }
    for (int fm = 0; fm < 2; ++fm)
      for (int fn = 0; fn < 4; ++fn)
        acc[fm][fn] = __builtin_amdgcn_wmma_f32_16x16x32_bf16(
            false, a[fm].v, false, b[fn].v, (short)0, acc[fm][fn], false, false);

    // write next tile to the other LDS buffer after this tile's ds_loads
    if (it + 1 < nk)
      stage_store<TA>(rg, As[cur ^ 1], Bs[cur ^ 1], tid);
    cur ^= 1;
  }

  // ---- epilogue: bias (+relu) + store (C/D layout: vgpr i -> M=lh*8+i) ----
  for (int fm = 0; fm < 2; ++fm)
    for (int fn = 0; fn < 4; ++fn)
      for (int i = 0; i < 8; ++i) {
        int row = bm0 + wm + fm * 16 + lh * 8 + i;
        int col = bn0 + wn + fn * 16 + l2;
        if (row < M && col < N) {
          float v = acc[fm][fn][i];
          if (bias) v += bias[col];
          if (relu && v < 0.f) v = 0.f;
          C[(long)row * ldc + col] = v;
        }
      }
}

// ---------------- small elementwise / reduction kernels --------------------
__global__ void gru_combine(float* __restrict__ Hio,
                            const float* Rx, const float* Rh,
                            const float* Zx, const float* Zh,
                            const float* Nx, const float* Nh, int n) {
  int i = blockIdx.x * blockDim.x + threadIdx.x;
  if (i >= n) return;
  float r = 1.f / (1.f + __expf(-(Rx[i] + Rh[i])));
  float z = 1.f / (1.f + __expf(-(Zx[i] + Zh[i])));
  float nn = tanhf(Nx[i] + r * Nh[i]);
  Hio[i] = (1.f - z) * nn + z * Hio[i];
}

__global__ void layernorm(float* __restrict__ X, const float* g, const float* b) {
  __shared__ float red[H];
  int r = blockIdx.x, t = threadIdx.x;
  float x = X[(long)r * H + t];
  red[t] = x; __syncthreads();
  for (int o = 64; o > 0; o >>= 1) { if (t < o) red[t] += red[t + o]; __syncthreads(); }
  float mu = red[0] / (float)H; __syncthreads();
  float d = x - mu;
  red[t] = d * d; __syncthreads();
  for (int o = 64; o > 0; o >>= 1) { if (t < o) red[t] += red[t + o]; __syncthreads(); }
  float var = red[0] / (float)H;
  X[(long)r * H + t] = d * rsqrtf(var + 1e-6f) * g[t] + b[t];
}

__global__ void concat2(const float* X, const float* Y, float* O, int rows) {
  int i = blockIdx.x * blockDim.x + threadIdx.x;
  if (i >= rows * 2 * H) return;
  int r = i >> 8, c = i & 255;
  O[i] = (c < H) ? X[(long)r * H + c] : Y[(long)r * H + (c - H)];
}

__global__ void build_vcat(const float* Nv, const float* assign, const float* svf,
                           float* O) {
  int i = blockIdx.x * blockDim.x + threadIdx.x;
  if (i >= V_N * VCK) return;
  int r = i / VCK, c = i % VCK;
  float v;
  if (c < H) v = Nv[(long)r * H + c];
  else if (c == H) v = assign[r];
  else v = svf[(long)r * FV + (c - H - 1)];
  O[i] = v;
}

// ---- edge-mask construction (rank-1 factors per agent) --------------------
__global__ void k_varmask(const int* av, float* vmask) {
  int i = blockIdx.x * blockDim.x + threadIdx.x;
  if (i >= A_N * M_N) return;
  int a = i / M_N, v = av[i];
  if (v >= 0) vmask[a * V_N + v] = 1.f;
}
__global__ void k_clausemask(const float* vmask, const float* Ap, const float* An,
                             float* cmask) {
  int i = blockIdx.x * blockDim.x + threadIdx.x;
  if (i >= A_N * C_N) return;
  int a = i / C_N, c = i % C_N;
  float s = 0.f;
  for (int v = 0; v < V_N; ++v)
    if (vmask[a * V_N + v] > 0.f &&
        (Ap[(long)v * C_N + c] + An[(long)v * C_N + c]) > 0.f) { s = 1.f; break; }
  cmask[i] = s;
}
__global__ void k_visible(const float* vmask, const float* cmask,
                          const float* Ap, const float* An, float* vis) {
  int i = blockIdx.x * blockDim.x + threadIdx.x;
  if (i >= A_N * V_N) return;
  int a = i / V_N, v = i % V_N;
  float s = (vmask[i] > 0.f) ? 1.f : 0.f;
  if (s == 0.f)
    for (int c = 0; c < C_N; ++c)
      if (cmask[a * C_N + c] > 0.f &&
          (Ap[(long)v * C_N + c] + An[(long)v * C_N + c]) > 0.f) { s = 1.f; break; }
  vis[i] = s;
}

// ---- actor head -----------------------------------------------------------
__global__ void actor_concat(const float* HpA, const float* HnA, const float* aid,
                             const int* av, float* X) {
  int i = blockIdx.x * blockDim.x + threadIdx.x;
  if (i >= A_N * M_N * XIN) return;
  int row = i / XIN, c = i % XIN;
  int a = row / M_N, m = row % M_N;
  int var = av[a * M_N + m]; if (var < 0) var = 0;
  float v;
  if (c < H)            v = HpA[((long)a * V_N + var) * H + c];
  else if (c < 2 * H)   v = HnA[((long)a * V_N + var) * H + (c - H)];
  else                  v = aid[a * E_N + (c - 2 * H)];
  X[(long)row * XIN + c] = v;
}
__global__ void actor_final(const float* H1, const float* W, const float* b,
                            const unsigned char* mask, float* out) {
  int i = blockIdx.x * blockDim.x + threadIdx.x;
  if (i >= A_N * M_N * 2) return;
  int r = i >> 1, j = i & 1;
  float s = b[j];
  for (int k = 0; k < 64; ++k) s += H1[(long)r * 64 + k] * W[k * 2 + j];
  out[i] = mask[r] ? s : -__builtin_inff();
}

// ---- critic head ----------------------------------------------------------
__global__ void pool_v(const float* Hp, const float* Hn, float* g) {
  __shared__ float ss[256], sm[256];
  int col = blockIdx.x, t = threadIdx.x;
  const float* X = (col < H) ? Hp : Hn; int c = col & (H - 1);
  float s = 0.f, mx = -__builtin_inff();
  for (int v = t; v < V_N; v += 256) { float x = X[(long)v * H + c]; s += x; mx = fmaxf(mx, x); }
  ss[t] = s; sm[t] = mx; __syncthreads();
  for (int o = 128; o > 0; o >>= 1) { if (t < o) { ss[t] += ss[t + o]; sm[t] = fmaxf(sm[t], sm[t + o]); } __syncthreads(); }
  if (t == 0) { g[col] = ss[0] / (float)V_N; g[256 + col] = sm[0]; }
}
__global__ void pool_c(const float* Hc, float* g) {
  __shared__ float ss[256], sm[256];
  int col = blockIdx.x, t = threadIdx.x;
  float s = 0.f, mx = -__builtin_inff();
  for (int c = t; c < C_N; c += 256) { float x = Hc[(long)c * H + col]; s += x; mx = fmaxf(mx, x); }
  ss[t] = s; sm[t] = mx; __syncthreads();
  for (int o = 128; o > 0; o >>= 1) { if (t < o) { ss[t] += ss[t + o]; sm[t] = fmaxf(sm[t], sm[t + o]); } __syncthreads(); }
  if (t == 0) { g[512 + col] = ss[0] / (float)C_N; g[640 + col] = sm[0]; }
}
__global__ void dense_row(const float* x, const float* W, const float* b,
                          float* y, int K, int N, int relu) {
  int n = blockIdx.x * blockDim.x + threadIdx.x;
  if (n >= N) return;
  float s = b ? b[n] : 0.f;
  for (int k = 0; k < K; ++k) s += x[k] * W[(long)k * N + n];
  if (relu && s < 0.f) s = 0.f;
  y[n] = s;
}

// ---------------- host-side orchestration ----------------------------------
static inline void gemm(hipStream_t s, int ta, const float* A, const float* B,
                        const float* bias, float* C, int M, int N, int K,
                        int lda, int ldb, int ldc,
                        const float* mm, const float* km, int relu) {
  dim3 g((M + BM - 1) / BM, (N + BN - 1) / BN), blk(256);
  int masked = (mm != 0);
  if (ta) {
    if (masked) gemm_k<1,1><<<g, blk, 0, s>>>(A, B, bias, C, M, N, K, lda, ldb, ldc, mm, km, relu);
    else        gemm_k<1,0><<<g, blk, 0, s>>>(A, B, bias, C, M, N, K, lda, ldb, ldc, mm, km, relu);
  } else {
    if (masked) gemm_k<0,1><<<g, blk, 0, s>>>(A, B, bias, C, M, N, K, lda, ldb, ldc, mm, km, relu);
    else        gemm_k<0,0><<<g, blk, 0, s>>>(A, B, bias, C, M, N, K, lda, ldb, ldc, mm, km, relu);
  }
}

struct GruP { const float *Whn,*Whr,*Whz,*Win,*Wir,*Wiz,*bhn,*bin,*bir,*biz; };
struct Net {
  const float *Ap,*An,*svf,*assign,*cf,*lnG,*lnB;
  const float *litPW,*litPB,*litNW,*litNB,*clW,*clB;
  const float *pcpW,*pcpB,*pcnW,*pcnB,*pvpW,*pvpB,*pvnW,*pvnB;
  GruP gc, gvp, gvn;
};
struct Ws {
  float *Tp,*Tn,*Tvp,*Tvn,*Mp,*Mn,*xcat,*vcp,*vcn,*G[6];
};

static void encoder(hipStream_t s, const Net& P, const Ws& w,
                    const float* vis, const float* cm,
                    float* Hp, float* Hn, float* Hc) {
  const int TPB = 256;
  // input embeddings
  gemm(s, 0, P.svf, P.litPW, P.litPB, Hp, V_N, H, FV, FV, H, H, 0, 0, 0);
  gemm(s, 0, P.svf, P.litNW, P.litNB, Hn, V_N, H, FV, FV, H, H, 0, 0, 0);
  gemm(s, 0, P.cf,  P.clW,   P.clB,   Hc, C_N, H, FV, FV, H, H, 0, 0, 0);
  for (int t = 0; t < STEPS; ++t) {
    // clause update
    gemm(s, 0, Hp, P.pcpW, P.pcpB, w.Tp, V_N, H, H, H, H, H, 0, 0, 0);
    gemm(s, 0, Hn, P.pcnW, P.pcnB, w.Tn, V_N, H, H, H, H, H, 0, 0, 0);
    gemm(s, 1, P.Ap, w.Tp, 0, w.Mp, C_N, H, V_N, C_N, H, H, cm, vis, 0); // Ap^T
    gemm(s, 1, P.An, w.Tn, 0, w.Mn, C_N, H, V_N, C_N, H, H, cm, vis, 0);
    concat2<<<(C_N * 2 * H + TPB - 1) / TPB, TPB, 0, s>>>(w.Mp, w.Mn, w.xcat, C_N);
    gemm(s, 0, w.xcat, P.gc.Wir, P.gc.bir, w.G[0], C_N, H, 2*H, 2*H, H, H, 0, 0, 0);
    gemm(s, 0, w.xcat, P.gc.Wiz, P.gc.biz, w.G[1], C_N, H, 2*H, 2*H, H, H, 0, 0, 0);
    gemm(s, 0, w.xcat, P.gc.Win, P.gc.bin, w.G[2], C_N, H, 2*H, 2*H, H, H, 0, 0, 0);
    gemm(s, 0, Hc, P.gc.Whr, 0,        w.G[3], C_N, H, H, H, H, H, 0, 0, 0);
    gemm(s, 0, Hc, P.gc.Whz, 0,        w.G[4], C_N, H, H, H, H, H, 0, 0, 0);
    gemm(s, 0, Hc, P.gc.Whn, P.gc.bhn, w.G[5], C_N, H, H, H, H, H, 0, 0, 0);
    gru_combine<<<(C_N * H + TPB - 1) / TPB, TPB, 0, s>>>(Hc, w.G[0], w.G[3], w.G[1], w.G[4], w.G[2], w.G[5], C_N * H);
    layernorm<<<C_N, H, 0, s>>>(Hc, P.lnG + (3*t)*H, P.lnB + (3*t)*H);
    // variable update
    gemm(s, 0, Hc, P.pvpW, P.pvpB, w.Tvp, C_N, H, H, H, H, H, 0, 0, 0);
    gemm(s, 0, Hc, P.pvnW, P.pvnB, w.Tvn, C_N, H, H, H, H, H, 0, 0, 0);
    gemm(s, 0, P.Ap, w.Tvp, 0, w.Tp, V_N, H, C_N, C_N, H, H, vis, cm, 0);
    gemm(s, 0, P.An, w.Tvn, 0, w.Tn, V_N, H, C_N, C_N, H, H, vis, cm, 0);
    build_vcat<<<(V_N * VCK + TPB - 1) / TPB, TPB, 0, s>>>(w.Tp, P.assign, P.svf, w.vcp);
    build_vcat<<<(V_N * VCK + TPB - 1) / TPB, TPB, 0, s>>>(w.Tn, P.assign, P.svf, w.vcn);
    // gru_vp (in place on Hp)
    gemm(s, 0, w.vcp, P.gvp.Wir, P.gvp.bir, w.G[0], V_N, H, VCK, VCK, H, H, 0, 0, 0);
    gemm(s, 0, w.vcp, P.gvp.Wiz, P.gvp.biz, w.G[1], V_N, H, VCK, VCK, H, H, 0, 0, 0);
    gemm(s, 0, w.vcp, P.gvp.Win, P.gvp.bin, w.G[2], V_N, H, VCK, VCK, H, H, 0, 0, 0);
    gemm(s, 0, Hp, P.gvp.Whr, 0,         w.G[3], V_N, H, H, H, H, H, 0, 0, 0);
    gemm(s, 0, Hp, P.gvp.Whz, 0,         w.G[4], V_N, H, H, H, H, H, 0, 0, 0);
    gemm(s, 0, Hp, P.gvp.Whn, P.gvp.bhn, w.G[5], V_N, H, H, H, H, H, 0, 0, 0);
    gru_combine<<<(V_N * H + TPB - 1) / TPB, TPB, 0, s>>>(Hp, w.G[0], w.G[3], w.G[1], w.G[4], w.G[2], w.G[5], V_N * H);
    // gru_vn (in place on Hn)
    gemm(s, 0, w.vcn, P.gvn.Wir, P.gvn.bir, w.G[0], V_N, H, VCK, VCK, H, H, 0, 0, 0);
    gemm(s, 0, w.vcn, P.gvn.Wiz, P.gvn.biz, w.G[1], V_N, H, VCK, VCK, H, H, 0, 0, 0);
    gemm(s, 0, w.vcn, P.gvn.Win, P.gvn.bin, w.G[2], V_N, H, VCK, VCK, H, H, 0, 0, 0);
    gemm(s, 0, Hn, P.gvn.Whr, 0,         w.G[3], V_N, H, H, H, H, H, 0, 0, 0);
    gemm(s, 0, Hn, P.gvn.Whz, 0,         w.G[4], V_N, H, H, H, H, H, 0, 0, 0);
    gemm(s, 0, Hn, P.gvn.Whn, P.gvn.bhn, w.G[5], V_N, H, H, H, H, H, 0, 0, 0);
    gru_combine<<<(V_N * H + TPB - 1) / TPB, TPB, 0, s>>>(Hn, w.G[0], w.G[3], w.G[1], w.G[4], w.G[2], w.G[5], V_N * H);
    layernorm<<<V_N, H, 0, s>>>(Hp, P.lnG + (3*t+1)*H, P.lnB + (3*t+1)*H);
    layernorm<<<V_N, H, 0, s>>>(Hn, P.lnG + (3*t+2)*H, P.lnB + (3*t+2)*H);
  }
}

extern "C" void kernel_launch(void* const* d_in, const int* in_sizes, int n_in,
                              void* d_out, int out_size, void* d_ws, size_t ws_size,
                              hipStream_t stream) {
  (void)in_sizes; (void)n_in; (void)out_size; (void)ws_size;
  auto f = [&](int i) { return (const float*)d_in[i]; };
  // ---- inputs (setup_inputs order; params pytree-flattened, sorted keys) ----
  Net P;
  P.Ap = f(0); P.An = f(1); P.svf = f(2); P.assign = f(3); P.cf = f(4);
  const int* agent_vars = (const int*)d_in[5];
  const unsigned char* amask = (const unsigned char*)d_in[6];
  // 7:actor_d0.W 8:.b 9:actor_d1.W 10:.b 11:actor_out.W 12:.b 13:agent_id_emb
  // 14:clause.W 15:.b 16:critic_d0.W 17:.b 18:critic_d1.W 19:.b 20:critic_out.W 21:.b
  // 22..31 gru_c{Whn,Whr,Whz,Win,Wir,Wiz,bhn,bin,bir,biz} 32..41 gru_vn 42..51 gru_vp
  // 52:lit_neg.W 53:.b 54:lit_pos.W 55:.b 56:ln_b 57:ln_g
  // 58:phi_c_neg.W 59:.b 60:phi_c_pos.W 61:.b 62:phi_v_neg.W 63:.b 64:phi_v_pos.W 65:.b
  auto grup = [&](int i) {
    GruP g; g.Whn=f(i); g.Whr=f(i+1); g.Whz=f(i+2); g.Win=f(i+3); g.Wir=f(i+4);
    g.Wiz=f(i+5); g.bhn=f(i+6); g.bin=f(i+7); g.bir=f(i+8); g.biz=f(i+9); return g;
  };
  P.gc = grup(22); P.gvn = grup(32); P.gvp = grup(42);
  P.litNW=f(52); P.litNB=f(53); P.litPW=f(54); P.litPB=f(55);
  P.lnB=f(56); P.lnG=f(57);
  P.pcnW=f(58); P.pcnB=f(59); P.pcpW=f(60); P.pcpB=f(61);
  P.pvnW=f(62); P.pvnB=f(63); P.pvpW=f(64); P.pvpB=f(65);
  P.clW=f(14); P.clB=f(15);

  // ---- workspace carve-up ----
  float* base = (float*)d_ws; size_t off = 0;
  auto al = [&](size_t n) { float* p = base + off; off += n; return p; };
  float* vmask = al(A_N * V_N);
  float* cmask = al(A_N * C_N);
  float* vis   = al(A_N * V_N);
  float* HpA = al((size_t)A_N * V_N * H);
  float* HnA = al((size_t)A_N * V_N * H);
  float* HpG = al((size_t)V_N * H);
  float* HnG = al((size_t)V_N * H);
  float* HcG = al((size_t)C_N * H);
  float* HcS = al((size_t)C_N * H);
  Ws w;
  w.Tp = al((size_t)V_N * H);  w.Tn = al((size_t)V_N * H);
  w.Tvp = al((size_t)C_N * H); w.Tvn = al((size_t)C_N * H);
  w.Mp = al((size_t)C_N * H);  w.Mn = al((size_t)C_N * H);
  w.xcat = al((size_t)C_N * 2 * H);
  w.vcp = al((size_t)V_N * VCK); w.vcn = al((size_t)V_N * VCK);
  for (int i = 0; i < 6; ++i) w.G[i] = al((size_t)C_N * H);
  float* aX  = al((size_t)A_N * M_N * XIN);
  float* aH0 = al((size_t)A_N * M_N * H);
  float* aH1 = al((size_t)A_N * M_N * 64);
  float* g   = al(768);
  float* ch0 = al(128);
  float* ch1 = al(64);

  const int TPB = 256;
  // ---- rank-1 edge-mask factors per agent ----
  hipMemsetAsync(vmask, 0, A_N * V_N * sizeof(float), stream);
  k_varmask  <<<(A_N*M_N + TPB-1)/TPB, TPB, 0, stream>>>(agent_vars, vmask);
  k_clausemask<<<(A_N*C_N + TPB-1)/TPB, TPB, 0, stream>>>(vmask, P.Ap, P.An, cmask);
  k_visible  <<<(A_N*V_N + TPB-1)/TPB, TPB, 0, stream>>>(vmask, cmask, P.Ap, P.An, vis);

  // ---- 4 masked encoders (actor) + 1 global encoder (critic) ----
  for (int a = 0; a < A_N; ++a)
    encoder(stream, P, w, vis + (size_t)a * V_N, cmask + (size_t)a * C_N,
            HpA + (size_t)a * V_N * H, HnA + (size_t)a * V_N * H, HcS);
  encoder(stream, P, w, 0, 0, HpG, HnG, HcG);

  // ---- actor head ----
  actor_concat<<<(A_N*M_N*XIN + TPB-1)/TPB, TPB, 0, stream>>>(HpA, HnA, f(13), agent_vars, aX);
  gemm(stream, 0, aX,  f(7), f(8),  aH0, A_N*M_N, H,  XIN, XIN, H,  H,  0, 0, 1);
  gemm(stream, 0, aH0, f(9), f(10), aH1, A_N*M_N, 64, H,   H,   64, 64, 0, 0, 1);
  actor_final<<<(A_N*M_N*2 + TPB-1)/TPB, TPB, 0, stream>>>(aH1, f(11), f(12), amask, (float*)d_out);

  // ---- critic head ----
  pool_v<<<256, 256, 0, stream>>>(HpG, HnG, g);
  pool_c<<<128, 256, 0, stream>>>(HcG, g);
  dense_row<<<1, 128, 0, stream>>>(g,   f(16), f(17), ch0, 6*H, 128, 1);
  dense_row<<<1, 64, 0, stream>>>(ch0, f(18), f(19), ch1, 128, 64, 1);
  dense_row<<<1, 1, 0, stream>>>(ch1, f(20), f(21), ((float*)d_out) + A_N*M_N*2, 64, 1, 0);
}
